// LSTM_37237366456429
// MI455X (gfx1250) — matrix operand order repaired and checked by reading, not compiled, not run on hardware
//
#include <hip/hip_runtime.h>
#include <hip/hip_bf16.h>
#include <math.h>

typedef __attribute__((ext_vector_type(16))) __bf16 v16bf;
typedef __attribute__((ext_vector_type(8)))  float  v8f;

#define TB   16      // batch rows per workgroup (WMMA M)
#define HH   128     // hidden size
#define KTOT 256     // combined K = D + H (both 128 here)
#define KS   264     // padded LDS K-stride (elements) to spread banks
#define NG   512     // 4*H gate rows of W

union Frag { v16bf v; uint4 q[2]; };

__device__ __forceinline__ float fast_tanh(float x) {
#if __has_builtin(__builtin_amdgcn_tanhf)
    return __builtin_amdgcn_tanhf(x);
#else
    return tanhf(x);
#endif
}
// sigmoid(x) = 0.5 + 0.5*tanh(0.5*x): one native trans op instead of exp+rcp
__device__ __forceinline__ float fast_sigmoid(float x) {
    return __builtin_fmaf(0.5f, fast_tanh(0.5f * x), 0.5f);
}

// ---------------------------------------------------------------------------
// Recurrent layer: one workgroup owns a 16-row batch tile for all T steps.
// Weights live entirely in VGPRs (loaded once); LDS holds only the
// double-buffered A panel [x_t | h_prev] in bf16. x(t+1) arrives via the
// CDNA5 async-to-LDS path (ASYNCcnt); c stays in VGPRs, h is shared across waves via LDS.
// ---------------------------------------------------------------------------
__global__ __launch_bounds__(256) void lstm_layer_kernel(
    const __bf16* __restrict__ xb,    // [B,T,128] bf16 input sequence
    const __bf16* __restrict__ Wb,    // [512][256] bf16 combined [Wih | Whh]
    const float*  __restrict__ b_ih,  // [512]
    const float*  __restrict__ b_hh,  // [512]
    __bf16* __restrict__ hseq_out,    // [B,T,128] bf16  (storeAll==1)
    float*  __restrict__ last_out,    // [B,128]   f32   (storeAll==0, t==T-1)
    int T, int storeAll)
{
    __shared__ __bf16 aPanel[2][TB * KS];     // 2 * 16 * 264 * 2B = 16896 B

    const int tid  = threadIdx.x;
    const int wave = tid >> 5;
    const int lane = tid & 31;
    const int lg   = lane >> 4;               // K-half select (ISA 7.12.2)
    const int ln   = lane & 15;
    const int b0   = blockIdx.x * TB;
    const int nc   = wave * 16 + ln;          // this lane's h column (0..127)

    // ---- B fragments: 4 gates x 8 K-tiles, register-resident for all T ----
    Frag B[4][8];
#pragma unroll
    for (int g = 0; g < 4; ++g) {
        const uint4* wrow = (const uint4*)(Wb + (size_t)(nc + g * 128) * KTOT);
#pragma unroll
        for (int kt = 0; kt < 8; ++kt) {
            B[g][kt].q[0] = wrow[kt * 4 + lg];         // dwords kt*16+4*lg   .. +3
            B[g][kt].q[1] = wrow[kt * 4 + 2 + lg];     // dwords kt*16+8+4*lg .. +3
        }
    }

    // ---- prologue: x(0) -> panel 0, h(0)=0 ----
    const int prow = tid >> 4;                // 0..15
    const int pch  = tid & 15;                // 16B chunk within 256B row
    {
        const uint4* xs = (const uint4*)(xb + ((long long)(b0 + prow) * T) * HH) + pch;
        uint4* dx = (uint4*)(&aPanel[0][prow * KS]) + pch;
        uint4* dh = (uint4*)(&aPanel[0][prow * KS + HH]) + pch;
        uint4  z  = make_uint4(0u, 0u, 0u, 0u);
        *dx = *xs;
        *dh = z;
    }

    // LDS byte offsets for this thread's async x chunk, per panel
    unsigned ldsoff[2];
    ldsoff[0] = (unsigned)(uintptr_t)((char*)&aPanel[0][prow * KS] + pch * 16);
    ldsoff[1] = (unsigned)(uintptr_t)((char*)&aPanel[1][prow * KS] + pch * 16);
    // global source pointer for x(1); advances by 256B (one timestep) per iter
    const char* gx = (const char*)(xb + ((long long)(b0 + prow) * T + 1) * HH) + pch * 16;

    const float bias_i = b_ih[nc]       + b_hh[nc];
    const float bias_f = b_ih[nc + 128] + b_hh[nc + 128];
    const float bias_g = b_ih[nc + 256] + b_hh[nc + 256];
    const float bias_o = b_ih[nc + 384] + b_hh[nc + 384];

    v8f cst;
#pragma unroll
    for (int v = 0; v < 8; ++v) cst[v] = 0.0f;

    __syncthreads();

    for (int t = 0; t < T; ++t) {
        const int cur = t & 1;
        const __bf16* aCur = aPanel[cur];
        __bf16*       aNxt = aPanel[1 - cur];

        // ---- async prefetch x(t+1) -> other panel (16B per thread) ----
        if (t + 1 < T) {
            unsigned long long ga = (unsigned long long)(uintptr_t)gx;
            asm volatile("global_load_async_to_lds_b128 %0, %1, off"
                         :: "v"(ldsoff[1 - cur]), "v"(ga) : "memory");
            gx += 2 * HH;   // 256 bytes = one timestep
        }

        // ---- gates = [x_t | h_prev] @ [Wih|Whh]^T + bias ----
        v8f ai, af, ag, ao;
#pragma unroll
        for (int v = 0; v < 8; ++v) { ai[v] = bias_i; af[v] = bias_f; ag[v] = bias_g; ao[v] = bias_o; }

        const uint4* arow = (const uint4*)(aCur + ln * KS);
#pragma unroll
        for (int kt = 0; kt < 8; ++kt) {
            Frag A;
            A.q[0] = arow[kt * 4 + lg];        // dwords kt*16+4*lg   .. +3
            A.q[1] = arow[kt * 4 + 2 + lg];    // dwords kt*16+8+4*lg .. +3
            ai = __builtin_amdgcn_wmma_f32_16x16x32_bf16(false, A.v, false, B[0][kt].v, (short)0, ai, false, false);
            af = __builtin_amdgcn_wmma_f32_16x16x32_bf16(false, A.v, false, B[1][kt].v, (short)0, af, false, false);
            ag = __builtin_amdgcn_wmma_f32_16x16x32_bf16(false, A.v, false, B[2][kt].v, (short)0, ag, false, false);
            ao = __builtin_amdgcn_wmma_f32_16x16x32_bf16(false, A.v, false, B[3][kt].v, (short)0, ao, false, false);
        }

        // ---- lane-local cell update (C/D layout: lane=col N, VGPR v -> M=v+8*lg) ----
        v8f hn;
#pragma unroll
        for (int v = 0; v < 8; ++v) {
            float iv = fast_sigmoid(ai[v]);
            float fv = fast_sigmoid(af[v]);
            float gv = fast_tanh(ag[v]);
            float ov = fast_sigmoid(ao[v]);
            float cv = fv * cst[v] + iv * gv;
            cst[v] = cv;
            hn[v]  = ov * fast_tanh(cv);
        }

        // h_new -> next A panel (bf16) and global output
#pragma unroll
        for (int v = 0; v < 8; ++v) {
            int m = v + 8 * lg;
            aNxt[m * KS + HH + nc] = (__bf16)hn[v];
        }
        if (storeAll) {
#pragma unroll
            for (int v = 0; v < 8; ++v) {
                int m = v + 8 * lg;
                hseq_out[((long long)(b0 + m) * T + t) * HH + nc] = (__bf16)hn[v];
            }
        } else if (t == T - 1) {
#pragma unroll
            for (int v = 0; v < 8; ++v) {
                int m = v + 8 * lg;
                last_out[(b0 + m) * HH + nc] = hn[v];
            }
        }

        asm volatile("s_wait_asynccnt 0" ::: "memory");   // async x writes visible
        __syncthreads();                                   // flip double buffer
    }
}

// ---- f32 -> bf16 streaming convert (4 elems/thread) ----
__global__ __launch_bounds__(256) void cvt_x_kernel(const float* __restrict__ x,
                                                    __bf16* __restrict__ xb, long long n)
{
    long long i = ((long long)blockIdx.x * blockDim.x + threadIdx.x) * 4;
    if (i < n) {
        float4 f = *(const float4*)(x + i);
        union { __bf16 h[4]; uint2 u; } p;
        p.h[0] = (__bf16)f.x; p.h[1] = (__bf16)f.y; p.h[2] = (__bf16)f.z; p.h[3] = (__bf16)f.w;
        *(uint2*)(xb + i) = p.u;
    }
}

// ---- build combined bf16 weight image Wb[512][256] = [Wih | Whh] ----
__global__ __launch_bounds__(256) void cvt_w_kernel(const float* __restrict__ W_ih,
                                                    const float* __restrict__ W_hh,
                                                    __bf16* __restrict__ Wb)
{
    int idx = blockIdx.x * blockDim.x + threadIdx.x;   // 0 .. 512*256-1
    int n = idx >> 8;
    int k = idx & 255;
    float v = (k < HH) ? W_ih[n * HH + k] : W_hh[n * HH + (k - HH)];
    Wb[n * KTOT + k] = (__bf16)v;
}

// out[b] = dot(last[b,:], W_fc[0,:]) + b_fc[0]   (C == 1)
__global__ void fc_kernel(const float* __restrict__ last, const float* __restrict__ W_fc,
                          const float* __restrict__ b_fc, float* __restrict__ out, int B)
{
    int b = blockIdx.x * blockDim.x + threadIdx.x;
    if (b < B) {
        float s = b_fc[0];
#pragma unroll 4
        for (int n = 0; n < HH; ++n) s += last[b * HH + n] * W_fc[n];
        out[b] = s;
    }
}

extern "C" void kernel_launch(void* const* d_in, const int* in_sizes, int n_in,
                              void* d_out, int out_size, void* d_ws, size_t ws_size,
                              hipStream_t stream)
{
    (void)in_sizes; (void)n_in; (void)out_size; (void)ws_size;
    const int B = 256, T = 512;

    const float* x     = (const float*)d_in[0];
    const float* W_ih0 = (const float*)d_in[1];
    const float* W_hh0 = (const float*)d_in[2];
    const float* b_ih0 = (const float*)d_in[3];
    const float* b_hh0 = (const float*)d_in[4];
    const float* W_ih1 = (const float*)d_in[5];
    const float* W_hh1 = (const float*)d_in[6];
    const float* b_ih1 = (const float*)d_in[7];
    const float* b_hh1 = (const float*)d_in[8];
    const float* W_fc  = (const float*)d_in[9];
    const float* b_fc  = (const float*)d_in[10];

    // workspace layout
    const long long nx = (long long)B * T * HH;              // 16,777,216 elems
    char* ws = (char*)d_ws;
    __bf16* xb   = (__bf16*)(ws);                            // 32 MB
    __bf16* h0b  = (__bf16*)(ws + nx * 2);                   // 32 MB
    __bf16* Wb0  = (__bf16*)(ws + nx * 4);                   // 256 KB
    __bf16* Wb1  = (__bf16*)(ws + nx * 4 + (size_t)NG * KTOT * 2);
    float*  last = (float*) (ws + nx * 4 + (size_t)NG * KTOT * 4);

    cvt_x_kernel<<<(int)(nx / 4 / 256), 256, 0, stream>>>(x, xb, nx);
    cvt_w_kernel<<<NG * KTOT / 256, 256, 0, stream>>>(W_ih0, W_hh0, Wb0);
    cvt_w_kernel<<<NG * KTOT / 256, 256, 0, stream>>>(W_ih1, W_hh1, Wb1);

    lstm_layer_kernel<<<B / TB, 256, 0, stream>>>(xb,  Wb0, b_ih0, b_hh0, h0b, last, T, 1);
    lstm_layer_kernel<<<B / TB, 256, 0, stream>>>(h0b, Wb1, b_ih1, b_hh1, h0b, last, T, 0);

    fc_kernel<<<1, 256, 0, stream>>>(last, W_fc, b_fc, (float*)d_out, B);
}
